// SegTransformerDecoder_34076270527054
// MI455X (gfx1250) — matrix operand order repaired and checked by reading, not compiled, not run on hardware
//
#include <hip/hip_runtime.h>
#include <hip/hip_bf16.h>
#include <math.h>

// ---------------------------------------------------------------------------
// SegTransformerDecoder for MI455X (gfx1250, wave32, WMMA).
// All conv layers lowered to 16x16-tile GEMMs on v_wmma_f32_16x16x32_f16,
// with compile-time-specialized shapes so the K/tap loops unroll and the
// scalar side constant-folds.
// ---------------------------------------------------------------------------

typedef __attribute__((ext_vector_type(16))) _Float16 v16h;
typedef __attribute__((ext_vector_type(8)))  float    v8f;

#define QTOT 4096   // 64*64 queries
#define DEMB 128
#define WB   64

#define ACT_NONE    0
#define ACT_GELU    1
#define ACT_SIGMOID 2

// ---------------------------------------------------------------------------
// Conv-as-GEMM kernel, compile-time specialized (KS = 1 or 3, pad = KS/2).
//   in  : [Q, CIN]  row-major (query-major)
//   w   : [COUT, CIN, KS, KS]
//   out : [Q, COUT]
// Block = 128 threads = 4 waves. Block tile: 64 queries x 16 out-channels.
// Each wave owns one 16(query)x16(channel) WMMA tile; the B (weight) tile is
// staged once per K-chunk in LDS and reused by all 4 waves.
// ---------------------------------------------------------------------------
template<int CIN, int COUT, int KS, int ACT>
__global__ __launch_bounds__(128)
void conv_wmma_kernel(const float* __restrict__ in, const float* __restrict__ w,
                      const float* __restrict__ bias, float* __restrict__ out)
{
    __shared__ _Float16 As[64][40];   // 64 queries x K(32) chunk (+pad)
    __shared__ _Float16 Bs[32][24];   // K(32) x 16 out-channels (+pad)

    const int tid  = threadIdx.x;
    const int lane = tid & 31;
    const int wv   = tid >> 5;            // wave id: which 16-query subtile
    const int tq   = blockIdx.x * 64;     // query tile base
    const int n0   = blockIdx.y * 16;     // out-channel tile base
    constexpr int RAD = KS / 2;

    v8f acc = {};

    for (int ky = 0; ky < KS; ++ky) {
        for (int kx = 0; kx < KS; ++kx) {
            const int dy = ky - RAD, dx = kx - RAD;
            for (int k0 = 0; k0 < CIN; k0 += 32) {
                // prefetch next K-chunk while this one is consumed
                if (k0 + 32 < CIN) {
                    __builtin_prefetch(&in[(size_t)tq * CIN + k0 + 32], 0, 0);
                    __builtin_prefetch(&w[((size_t)n0 * CIN + k0 + 32) * KS * KS], 0, 0);
                }
                // ---- stage A tile: 64 queries x 32 input channels (shifted tap)
                for (int i = tid; i < 64 * 32; i += 128) {
                    const int m  = i >> 5, kk = i & 31;
                    const int q  = tq + m;
                    const int y  = q >> 6, x = q & 63;
                    const int yy = y + dy, xx = x + dx;
                    float v = 0.f;
                    if (yy >= 0 && yy < WB && xx >= 0 && xx < WB)
                        v = in[(size_t)(yy * WB + xx) * CIN + k0 + kk];
                    As[m][kk] = (_Float16)v;
                }
                // ---- stage B tile: 32 input channels x 16 out channels
                for (int i = tid; i < 32 * 16; i += 128) {
                    const int kk = i >> 4, n = i & 15;
                    const int co = n0 + n;
                    float v = 0.f;
                    if ((COUT % 16 == 0) || (co < COUT))
                        v = w[((size_t)co * CIN + k0 + kk) * KS * KS + ky * KS + kx];
                    Bs[kk][n] = (_Float16)v;
                }
                __syncthreads();

                // ---- pack per-lane fragments (ISA 7.12.2 16-bit layouts)
                // A: lane m = lane&15, K half = (lane>>4)*8; element e -> K
                //    k = (e>>3)*16 + 8*half + (e&7)
                // B: lane n = lane&15; element e -> k = (e>>3)*16 + 8*half + (e&7)
                const int mrow = lane & 15;
                const int half = lane >> 4;
                v16h a, b;
#pragma unroll
                for (int e = 0; e < 16; ++e)
                    a[e] = As[(wv << 4) + mrow][((e >> 3) << 4) + (half << 3) + (e & 7)];
#pragma unroll
                for (int e = 0; e < 16; ++e)
                    b[e] = Bs[((e >> 3) << 4) + (half << 3) + (e & 7)][mrow];

                acc = __builtin_amdgcn_wmma_f32_16x16x32_f16(
                        false, a, false, b, (short)0, acc, false, false);
                __syncthreads();
            }
        }
    }

    // ---- epilogue: D layout -> lane n = lane&15, rows m = e + 8*(lane>>4)
    const int n  = lane & 15;
    const int hh = lane >> 4;
    const int co = n0 + n;
    if ((COUT % 16 == 0) || (co < COUT)) {
        const float bv = bias[co];
#pragma unroll
        for (int e = 0; e < 8; ++e) {
            const int m = (wv << 4) + (hh << 3) + e;
            const int q = tq + m;
            float v = acc[e] + bv;
            if (ACT == ACT_GELU)
                v = 0.5f * v * (1.f + erff(v * 0.70710678118654752440f));
            else if (ACT == ACT_SIGMOID)
                v = 1.f / (1.f + __expf(-v));
            out[(size_t)q * COUT + co] = v;
        }
    }
}

// ---------------------------------------------------------------------------
// q = InstanceNorm(q + x), per channel over all Q (biased variance, eps 1e-5).
// One block (256 threads) per channel; q updated in place.
// ---------------------------------------------------------------------------
__global__ __launch_bounds__(256)
void addnorm_kernel(float* __restrict__ q, const float* __restrict__ x, int C)
{
    const int c = blockIdx.x;
    const int t = threadIdx.x;
    __shared__ float red[256];

    float s = 0.f, s2 = 0.f;
    for (int i = t; i < QTOT; i += 256) {
        const float v = q[(size_t)i * C + c] + x[(size_t)i * C + c];
        s += v; s2 += v * v;
    }
    red[t] = s; __syncthreads();
    for (int o = 128; o > 0; o >>= 1) { if (t < o) red[t] += red[t + o]; __syncthreads(); }
    const float mean = red[0] * (1.f / QTOT);
    __syncthreads();
    red[t] = s2; __syncthreads();
    for (int o = 128; o > 0; o >>= 1) { if (t < o) red[t] += red[t + o]; __syncthreads(); }
    const float var = red[0] * (1.f / QTOT) - mean * mean;
    const float rs  = rsqrtf(var + 1e-5f);
    for (int i = t; i < QTOT; i += 256) {
        const float v = q[(size_t)i * C + c] + x[(size_t)i * C + c];
        q[(size_t)i * C + c] = (v - mean) * rs;
    }
}

// ---------------------------------------------------------------------------
// Positional encoding MLP: pe = relu(pos2 @ w1 + b1) @ w2 + b2
//   pos2 = bev_pos[0, 0, y, x, 0:2]; w1:[2,256] w2:[256,128]
// One block per query, 128 threads.
// ---------------------------------------------------------------------------
__global__ __launch_bounds__(128)
void pe_kernel(const float* __restrict__ bev_pos,
               const float* __restrict__ w1, const float* __restrict__ b1,
               const float* __restrict__ w2, const float* __restrict__ b2,
               float* __restrict__ pe)
{
    const int q = blockIdx.x, t = threadIdx.x;
    const int y = q >> 6, x = q & 63;
    const float px = bev_pos[((size_t)(0 * WB + y) * WB + x) * 3 + 0];
    const float py = bev_pos[((size_t)(0 * WB + y) * WB + x) * 3 + 1];
    __shared__ float hid[256];
    for (int j = t; j < 256; j += 128) {
        float h = px * w1[j] + py * w1[256 + j] + b1[j];
        hid[j] = h > 0.f ? h : 0.f;
    }
    __syncthreads();
    float acc = b2[t];
    for (int k = 0; k < 256; ++k) acc += hid[k] * w2[(size_t)k * DEMB + t];
    pe[(size_t)q * DEMB + t] = acc;
}

// ---------------------------------------------------------------------------
// Deformable multi-cam / multi-level sampler.
//   off : [Q,12]  (already sigmoid'ed)   ch = p*3 + d
//   swl : [Q,16]  (raw logits)           ch = p*4 + l ; softmax over l here
//   out : [Q,512] col = p*128 + c        ('b q g p c -> b (p g c) h w')
// Block = one query, 128 threads (c = channel).
// ---------------------------------------------------------------------------
__global__ __launch_bounds__(128)
void sampler_kernel(const float* __restrict__ off, const float* __restrict__ swl,
                    const float* __restrict__ bev_pos, const float* __restrict__ l2i,
                    const float* __restrict__ f0, const float* __restrict__ f1,
                    const float* __restrict__ f2, const float* __restrict__ f3,
                    float* __restrict__ out)
{
    const int q = blockIdx.x, c = threadIdx.x;
    const int y = q >> 6, x = q & 63;

    __shared__ float su[4][6], sv[4][6], sm[4][6];
    __shared__ float ssw[4][4];

    if (c < 24) {                         // 4 points x 6 cameras: project once
        const int p = c / 6, n = c % 6;
        const float lo[3]   = {-51.2f, -51.2f, -5.0f};
        const float span[3] = {102.4f, 102.4f,  8.0f};
        float pt[3];
        for (int d = 0; d < 3; ++d) {
            const float r = bev_pos[((size_t)(p * WB + y) * WB + x) * 3 + d] * span[d] + lo[d];
            pt[d] = r + (off[(size_t)q * 12 + p * 3 + d] - 0.5f);
        }
        const float* M = l2i + (size_t)n * 16;
        const float c0 = M[0]*pt[0] + M[1]*pt[1] + M[2]*pt[2]  + M[3];
        const float c1 = M[4]*pt[0] + M[5]*pt[1] + M[6]*pt[2]  + M[7];
        const float c2 = M[8]*pt[0] + M[9]*pt[1] + M[10]*pt[2] + M[11];
        const float den = fmaxf(c2, 1e-6f);
        const float u = c0 / den * (1.f / 480.f);
        const float v = c1 / den * (1.f / 224.f);
        su[p][n] = u; sv[p][n] = v;
        sm[p][n] = (c2 > 1e-6f && u > 0.f && u < 1.f && v > 0.f && v < 1.f) ? 1.f : 0.f;
    }
    if (c >= 64 && c < 68) {              // softmax over the 4 levels, per point
        const int p = c - 64;
        float lv[4], mx = -1e30f;
        for (int l = 0; l < 4; ++l) { lv[l] = swl[(size_t)q * 16 + p * 4 + l]; mx = fmaxf(mx, lv[l]); }
        float s = 0.f;
        for (int l = 0; l < 4; ++l) { lv[l] = __expf(lv[l] - mx); s += lv[l]; }
        const float inv = 1.f / s;
        for (int l = 0; l < 4; ++l) ssw[p][l] = lv[l] * inv;
    }
    __syncthreads();

    const float* fptr[4] = {f0, f1, f2, f3};
    const int    fh[4]   = {28, 14, 7, 4};
    const int    fw[4]   = {60, 30, 15, 8};

    for (int p = 0; p < 4; ++p) {
        float acc = 0.f;
        for (int l = 0; l < 4; ++l) {
            const float* F = fptr[l];
            const int H = fh[l], W = fw[l];
            float al = 0.f;
            for (int n = 0; n < 6; ++n) {
                if (sm[p][n] == 0.f) continue;
                const float ix = su[p][n] * W - 0.5f;
                const float iy = sv[p][n] * H - 0.5f;
                const float fx = floorf(ix), fy = floorf(iy);
                const int x0 = (int)fx, y0 = (int)fy;
                const float wx = ix - fx, wy = iy - fy;
                const float* base = F + ((size_t)n * DEMB + c) * H * W;
                auto g = [&](int yi, int xi) -> float {
                    if (yi < 0 || yi >= H || xi < 0 || xi >= W) return 0.f;
                    return base[(size_t)yi * W + xi];
                };
                al += g(y0, x0)     * (1.f - wx) * (1.f - wy)
                    + g(y0, x0 + 1) * wx         * (1.f - wy)
                    + g(y0 + 1, x0) * (1.f - wx) * wy
                    + g(y0 + 1, x0 + 1) * wx     * wy;
            }
            acc += al * ssw[p][l];
        }
        out[(size_t)q * 512 + p * DEMB + c] = acc;
    }
}

// ---------------------------------------------------------------------------
// Layout shuffles + residual add
// ---------------------------------------------------------------------------
__global__ void cq_to_qc_kernel(const float* __restrict__ in, float* __restrict__ out)
{   // [C,Q] -> [Q,C]
    const int idx = blockIdx.x * 256 + threadIdx.x;
    if (idx < DEMB * QTOT) {
        const int c = idx / QTOT, q = idx % QTOT;
        out[(size_t)q * DEMB + c] = in[idx];
    }
}
__global__ void qc_to_cq_kernel(const float* __restrict__ in, float* __restrict__ out)
{   // [Q,C] -> [C,Q]
    const int idx = blockIdx.x * 256 + threadIdx.x;
    if (idx < DEMB * QTOT) {
        const int q = idx / DEMB, c = idx % DEMB;
        out[(size_t)c * QTOT + q] = in[idx];
    }
}
__global__ void add_kernel(const float* __restrict__ a, const float* __restrict__ b,
                           float* __restrict__ o, int n)
{
    const int idx = blockIdx.x * 256 + threadIdx.x;
    if (idx < n) o[idx] = a[idx] + b[idx];
}

// ---------------------------------------------------------------------------
template<int CIN, int COUT, int KS, int ACT>
static inline void launch_conv(const float* in, const float* w, const float* b,
                               float* out, hipStream_t s)
{
    dim3 grid(QTOT / 64, (COUT + 15) / 16);
    conv_wmma_kernel<CIN, COUT, KS, ACT><<<grid, 128, 0, s>>>(in, w, b, out);
}

extern "C" void kernel_launch(void* const* d_in, const int* in_sizes, int n_in,
                              void* d_out, int out_size, void* d_ws, size_t ws_size,
                              hipStream_t stream)
{
    const float* feat0     = (const float*)d_in[0];
    const float* feat1     = (const float*)d_in[1];
    const float* feat2     = (const float*)d_in[2];
    const float* feat3     = (const float*)d_in[3];
    const float* lidar2img = (const float*)d_in[4];
    const float* bev_query = (const float*)d_in[5];
    const float* bev_pos   = (const float*)d_in[6];
    const float* pe_w1 = (const float*)d_in[7],  *pe_b1 = (const float*)d_in[8];
    const float* pe_w2 = (const float*)d_in[9],  *pe_b2 = (const float*)d_in[10];
    const float* ic_w1 = (const float*)d_in[11], *ic_b1 = (const float*)d_in[12];
    const float* ic_w2 = (const float*)d_in[13], *ic_b2 = (const float*)d_in[14];
    const float* ic_w3 = (const float*)d_in[15], *ic_b3 = (const float*)d_in[16];
    const float* mc_w1 = (const float*)d_in[17], *mc_b1 = (const float*)d_in[18];
    const float* mc_w2 = (const float*)d_in[19], *mc_b2 = (const float*)d_in[20];
    const float* mc_w3 = (const float*)d_in[21], *mc_b3 = (const float*)d_in[22];
    const float* oc_w1 = (const float*)d_in[23], *oc_b1 = (const float*)d_in[24];
    const float* oc_w2 = (const float*)d_in[25], *oc_b2 = (const float*)d_in[26];
    const float* oc_w3 = (const float*)d_in[27], *oc_b3 = (const float*)d_in[28];
    const float* so_w  = (const float*)d_in[29], *so_b  = (const float*)d_in[30];
    const float* sw_w  = (const float*)d_in[31], *sw_b  = (const float*)d_in[32];

    float* ws    = (float*)d_ws;
    float* QBUF  = ws;                       // [Q,128]
    float* PEBUF = QBUF  + (size_t)QTOT * DEMB;
    float* S512  = PEBUF + (size_t)QTOT * DEMB;   // [Q,512]
    float* T1    = S512  + (size_t)QTOT * 512;    // [Q,256] scratch
    float* T2    = T1    + (size_t)QTOT * 256;    // [Q,256] scratch
    float* OFFB  = T2    + (size_t)QTOT * 256;    // [Q,12]
    float* SWB   = OFFB  + (size_t)QTOT * 12;     // [Q,16]

    const int elt_blocks = (QTOT * DEMB + 255) / 256;

    // q <- bev_query transposed to query-major; pe computed once (shared weights)
    cq_to_qc_kernel<<<elt_blocks, 256, 0, stream>>>(bev_query, QBUF);
    pe_kernel<<<QTOT, 128, 0, stream>>>(bev_pos, pe_w1, pe_b1, pe_w2, pe_b2, PEBUF);

    for (int layer = 0; layer < 2; ++layer) {
        // ---- in_conv block: x = conv1(gelu) -> conv3(gelu) -> conv1 of (q+pe)
        add_kernel<<<elt_blocks, 256, 0, stream>>>(QBUF, PEBUF, T1, QTOT * DEMB);
        launch_conv<128, 128, 1, ACT_GELU>(T1, ic_w1, ic_b1, T2, stream);
        launch_conv<128, 128, 3, ACT_GELU>(T2, ic_w2, ic_b2, T1, stream);
        launch_conv<128, 128, 1, ACT_NONE>(T1, ic_w3, ic_b3, T2, stream);
        addnorm_kernel<<<DEMB, 256, 0, stream>>>(QBUF, T2, DEMB);

        // ---- sampling offsets (sigmoid) + scale-weight logits
        launch_conv<128, 12, 1, ACT_SIGMOID>(QBUF, so_w, so_b, OFFB, stream);
        launch_conv<128, 16, 1, ACT_NONE>(QBUF, sw_w, sw_b, SWB, stream);

        // ---- project, bilinear-sample 6 cams x 4 levels, weighted sum
        sampler_kernel<<<QTOT, 128, 0, stream>>>(OFFB, SWB, bev_pos, lidar2img,
                                                 feat0, feat1, feat2, feat3, S512);

        // ---- mid_conv block on sampled features
        launch_conv<512, 256, 1, ACT_GELU>(S512, mc_w1, mc_b1, T1, stream);
        launch_conv<256, 128, 3, ACT_GELU>(T1, mc_w2, mc_b2, T2, stream);
        launch_conv<128, 128, 1, ACT_NONE>(T2, mc_w3, mc_b3, T1, stream);
        addnorm_kernel<<<DEMB, 256, 0, stream>>>(QBUF, T1, DEMB);

        // ---- out_conv block
        launch_conv<128, 128, 1, ACT_GELU>(QBUF, oc_w1, oc_b1, T1, stream);
        launch_conv<128, 128, 3, ACT_GELU>(T1, oc_w2, oc_b2, T2, stream);
        launch_conv<128, 128, 1, ACT_NONE>(T2, oc_w3, oc_b3, T1, stream);
        addnorm_kernel<<<DEMB, 256, 0, stream>>>(QBUF, T1, DEMB);
    }

    // back to [B, D, H, W]
    qc_to_cq_kernel<<<elt_blocks, 256, 0, stream>>>(QBUF, (float*)d_out);
}
